// MultiHeadSelfAttention_13486197309863
// MI455X (gfx1250) — compile-verified, matrix operand order checked by reading, not run on hardware
//
#include <hip/hip_runtime.h>
#include <hip/hip_bf16.h>
#include <math.h>

typedef __bf16 bf16_t;
typedef bf16_t v16bf __attribute__((ext_vector_type(16)));
typedef bf16_t v8bf  __attribute__((ext_vector_type(8)));
typedef float  v8f   __attribute__((ext_vector_type(8)));

#define AS3 __attribute__((address_space(3)))

// Inline asm for CDNA5 async global->LDS copies (assembler-verified in round 3).
#define USE_ASYNC_ASM 1

constexpr int Bn = 2, Tn = 2048, Dn = 1024, Hn = 16, DHn = 64;
constexpr int Nrows = Bn * Tn;  // 4096
constexpr int LDSPAD = 72;      // padded row stride (elements) for 64-wide k slices
                                // 144B = 36 banks; 16 rows -> 16 distinct bank groups

__device__ __forceinline__ v8f wmma_bf16f32(v16bf a, v16bf b, v8f c) {
  return __builtin_amdgcn_wmma_f32_16x16x32_bf16(
      false, a, false, b, (short)0, c, false, false);
}

// ---- global fragment load (used by attention): 16x32 bf16 tile ----
// lane&15 = row (M for A, N-col==row of [N,K] weight for B); lane>>4 = K half.
__device__ __forceinline__ v16bf load_frag(const bf16_t* __restrict__ base, int ld,
                                           int row0, int k0, int lane) {
  const bf16_t* p = base + (size_t)(row0 + (lane & 15)) * ld + (k0 + ((lane >> 4) << 4));
  return *(const v16bf*)p;
}

// ---- LDS fragment load from a padded (LDSPAD-stride) tile, k-offset kk ----
__device__ __forceinline__ v16bf lds_frag(const bf16_t* buf, int row0, int kk, int lane) {
  const bf16_t* p = buf + (row0 + (lane & 15)) * LDSPAD + kk + ((lane >> 4) << 4);
  union { v16bf v; v8bf h[2]; } u;
  u.h[0] = *(const v8bf*)p;
  u.h[1] = *(const v8bf*)(p + 8);
  return u.v;
}

// ---- 16B global -> LDS copy (CDNA5 async; ASYNCcnt-tracked) ----
__device__ __forceinline__ void copy16_g2l(const bf16_t* g, bf16_t* l) {
#if USE_ASYNC_ASM
  AS3 bf16_t* l3 = (AS3 bf16_t*)l;                      // addrspacecast to LDS
  unsigned ldsoff = (unsigned)(unsigned long long)l3;   // 32-bit LDS byte addr
  unsigned long long gaddr = (unsigned long long)g;
  asm volatile("global_load_async_to_lds_b128 %0, %1, off"
               :: "v"(ldsoff), "v"(gaddr)
               : "memory");
#else
  *(v8bf*)l = *(const v8bf*)g;
#endif
}

__device__ __forceinline__ void wait_async_copies() {
#if USE_ASYNC_ASM
  asm volatile("s_wait_asynccnt 0x0" ::: "memory");
#endif
}

__global__ __launch_bounds__(256)
void f32_to_bf16_kernel(const float* __restrict__ src, bf16_t* __restrict__ dst, int n) {
  int i = blockIdx.x * blockDim.x + threadIdx.x;
  if (i < n) dst[i] = (bf16_t)src[i];
}

// C[M,N] = A[M,K] * W[N,K]^T, bf16 in, f32 accumulate.
// Block: 256 thr = 8 waves, tile 128(M) x 64(N), K-slice 64, double-buffered
// async global->LDS staging shared by all waves (8 WMMAs per barrier region).
// MODE 0: f32 -> outF[M,N].  MODE 1: qkv scatter epilogue.
template <int MODE>
__global__ __launch_bounds__(256)
void gemm_wmma(const bf16_t* __restrict__ A, const bf16_t* __restrict__ W,
               float* __restrict__ outF, bf16_t* __restrict__ qb,
               bf16_t* __restrict__ kb, bf16_t* __restrict__ vtb,
               int M, int N, int K) {
  __shared__ __align__(128) bf16_t As[2][128 * LDSPAD];  // 36 KB
  __shared__ __align__(128) bf16_t Bs[2][64 * LDSPAD];   // 18 KB
  (void)M;

  const int tid  = threadIdx.x;
  const int lane = tid & 31;
  const int wid  = tid >> 5;
  const int mBlock = blockIdx.y * 128;
  const int nBlock = blockIdx.x * 64;
  const int m0w = (wid >> 1) * 32;   // wave tile inside block (M)
  const int n0w = (wid & 1) * 32;    // wave tile inside block (N)

  // Stage one 64-wide K-slice (A:128x64, B:64x64) into LDS buffer `buf`.
  auto stage = [&](int buf, int k0) {
#pragma unroll
    for (int i = 0; i < 4; ++i) {            // A: 1024 x 16B chunks
      int c = tid + 256 * i;
      int row = c >> 3, ck = (c & 7) * 8;
      copy16_g2l(A + (size_t)(mBlock + row) * K + k0 + ck,
                 &As[buf][row * LDSPAD + ck]);
    }
#pragma unroll
    for (int i = 0; i < 2; ++i) {            // B: 512 x 16B chunks
      int c = tid + 256 * i;
      int row = c >> 3, ck = (c & 7) * 8;
      copy16_g2l(W + (size_t)(nBlock + row) * K + k0 + ck,
                 &Bs[buf][row * LDSPAD + ck]);
    }
  };

  v8f c[2][2] = {};
  const int nk = K / 64;
  stage(0, 0);
  for (int kt = 0; kt < nk; ++kt) {
    const int cur = kt & 1;
    wait_async_copies();        // this wave's copies into `cur` done
    __syncthreads();            // ... and everyone else's
    if (kt + 1 < nk) stage(cur ^ 1, (kt + 1) * 64);  // overlap next copy

#pragma unroll
    for (int kk = 0; kk < 64; kk += 32) {
      v16bf a0 = lds_frag(As[cur], m0w,      kk, lane);
      v16bf a1 = lds_frag(As[cur], m0w + 16, kk, lane);
      v16bf b0 = lds_frag(Bs[cur], n0w,      kk, lane);
      v16bf b1 = lds_frag(Bs[cur], n0w + 16, kk, lane);
      c[0][0] = wmma_bf16f32(a0, b0, c[0][0]);
      c[0][1] = wmma_bf16f32(a0, b1, c[0][1]);
      c[1][0] = wmma_bf16f32(a1, b0, c[1][0]);
      c[1][1] = wmma_bf16f32(a1, b1, c[1][1]);
    }

    __syncthreads();            // all waves done reading `cur` before reuse
  }

  const int half = lane >> 4, cl = lane & 15;
#pragma unroll
  for (int ti = 0; ti < 2; ++ti)
#pragma unroll
    for (int tj = 0; tj < 2; ++tj)
#pragma unroll
      for (int r = 0; r < 8; ++r) {
        int row = mBlock + m0w + 16 * ti + r + 8 * half;
        int col = nBlock + n0w + 16 * tj + cl;
        float v = c[ti][tj][r];
        if (MODE == 0) {
          outF[(size_t)row * N + col] = v;
        } else {
          int part = col >> 10;          // 0=q, 1=k, 2=v
          int d  = col & (Dn - 1);
          int h  = d >> 6;
          int dh = d & (DHn - 1);
          int b  = row >> 11;
          int t  = row & (Tn - 1);
          size_t bh = (size_t)b * Hn + h;
          if (part == 0)      qb [(bh * Tn + t) * DHn + dh] = (bf16_t)v;
          else if (part == 1) kb [(bh * Tn + t) * DHn + dh] = (bf16_t)v;
          else                vtb[(bh * DHn + dh) * Tn + t] = (bf16_t)v;
        }
      }
}

// Interleaved-pair RoPE on q and k, in place. One thread per (b,h,t, pair i<32).
__global__ __launch_bounds__(256)
void rope_kernel(bf16_t* __restrict__ qb, bf16_t* __restrict__ kb) {
  int idx = blockIdx.x * blockDim.x + threadIdx.x;
  int i  = idx & 31;
  int t  = (idx >> 5) & (Tn - 1);
  int bh = idx >> 16;
  if (bh >= Bn * Hn) return;
  float freq = __expf(-0.2878231366242557f * (float)i);  // ln(1e4)*2/64
  float ang = (float)t * freq;
  float sn, cs;
  __sincosf(ang, &sn, &cs);
  size_t base = ((size_t)bh * Tn + t) * DHn + 2 * i;
  {
    float x0 = (float)qb[base], x1 = (float)qb[base + 1];
    qb[base]     = (bf16_t)(cs * x0 - sn * x1);
    qb[base + 1] = (bf16_t)(cs * x1 + sn * x0);
  }
  {
    float x0 = (float)kb[base], x1 = (float)kb[base + 1];
    kb[base]     = (bf16_t)(cs * x0 - sn * x1);
    kb[base + 1] = (bf16_t)(cs * x1 + sn * x0);
  }
}

// Flash attention: one wave per 16-query tile of one (b,h). Online softmax.
// Causal masking only evaluated on diagonal chunks (wave-uniform branch).
__global__ __launch_bounds__(256)
void flash_attn_wmma(const bf16_t* __restrict__ qb, const bf16_t* __restrict__ kb,
                     const bf16_t* __restrict__ vtb, bf16_t* __restrict__ attnb) {
  __shared__ __align__(32) bf16_t pbuf_all[8][16 * 32];
  const int lane = threadIdx.x & 31;
  const int wid  = threadIdx.x >> 5;
  const int w  = blockIdx.x * 8 + wid;
  const int bh = w >> 7;
  const int qt = w & 127;
  const int q0 = qt * 16;
  const int b = bh >> 4, h = bh & (Hn - 1);

  const bf16_t* Q  = qb  + (size_t)bh * Tn * DHn;
  const bf16_t* Kp = kb  + (size_t)bh * Tn * DHn;
  const bf16_t* Vt = vtb + (size_t)bh * DHn * Tn;   // [DH, T]
  bf16_t* pb = pbuf_all[wid];

  const int half = lane >> 4, cl = lane & 15;

  v16bf aq0 = load_frag(Q, DHn, q0, 0, lane);
  v16bf aq1 = load_frag(Q, DHn, q0, 32, lane);

  v8f o[4] = {};
  float mrow[8], lrow[8];
#pragma unroll
  for (int r = 0; r < 8; ++r) { mrow[r] = -__builtin_huge_valf(); lrow[r] = 0.f; }

  const float scale = 0.125f;  // 1/sqrt(64)
  for (int k0 = 0; k0 <= q0; k0 += 32) {
    const bool diag = (k0 + 31) > q0;   // wave-uniform: chunk touches diagonal

    v16bf bkA = load_frag(Kp, DHn, k0,      0,  lane);
    v16bf bkB = load_frag(Kp, DHn, k0,      32, lane);
    v16bf bkC = load_frag(Kp, DHn, k0 + 16, 0,  lane);
    v16bf bkD = load_frag(Kp, DHn, k0 + 16, 32, lane);
    v8f s0 = {}, s1 = {};
    s0 = wmma_bf16f32(aq0, bkA, s0);
    s0 = wmma_bf16f32(aq1, bkB, s0);
    s1 = wmma_bf16f32(aq0, bkC, s1);
    s1 = wmma_bf16f32(aq1, bkD, s1);

#pragma unroll
    for (int r = 0; r < 8; ++r) {
      int rowg = q0 + r + 8 * half;
      float v0 = s0[r] * scale;
      float v1 = s1[r] * scale;
      if (diag) {
        if (k0 + cl > rowg)      v0 = -__builtin_huge_valf();
        if (k0 + 16 + cl > rowg) v1 = -__builtin_huge_valf();
      }
      float mx = fmaxf(v0, v1);
#pragma unroll
      for (int off = 1; off < 16; off <<= 1) mx = fmaxf(mx, __shfl_xor(mx, off, 16));
      float mnew = fmaxf(mrow[r], mx);
      float corr = __expf(mrow[r] - mnew);
      float p0 = __expf(v0 - mnew);
      float p1 = __expf(v1 - mnew);
      float ps = p0 + p1;
#pragma unroll
      for (int off = 1; off < 16; off <<= 1) ps += __shfl_xor(ps, off, 16);
      lrow[r] = lrow[r] * corr + ps;
      mrow[r] = mnew;
#pragma unroll
      for (int j = 0; j < 4; ++j) o[j][r] *= corr;
      pb[(r + 8 * half) * 32 + cl]      = (bf16_t)p0;
      pb[(r + 8 * half) * 32 + 16 + cl] = (bf16_t)p1;
    }

    v16bf ap = *(const v16bf*)&pb[(lane & 15) * 32 + ((lane >> 4) << 4)];
#pragma unroll
    for (int j = 0; j < 4; ++j) {
      v16bf bv = load_frag(Vt, Tn, 16 * j, k0, lane);
      o[j] = wmma_bf16f32(ap, bv, o[j]);
    }
  }

#pragma unroll
  for (int r = 0; r < 8; ++r) {
    float inv = 1.f / lrow[r];
    int t = q0 + r + 8 * half;
    size_t rowbase = ((size_t)(b * Tn + t)) * Dn + (size_t)h * DHn;
#pragma unroll
    for (int j = 0; j < 4; ++j)
      attnb[rowbase + 16 * j + cl] = (bf16_t)(o[j][r] * inv);
  }
}

extern "C" void kernel_launch(void* const* d_in, const int* in_sizes, int n_in,
                              void* d_out, int out_size, void* d_ws, size_t ws_size,
                              hipStream_t stream) {
  (void)in_sizes; (void)n_in; (void)out_size; (void)ws_size;
  const float* x    = (const float*)d_in[0];
  const float* Wqkv = (const float*)d_in[1];
  const float* Wo   = (const float*)d_in[2];
  float* out = (float*)d_out;

  char* ws = (char*)d_ws;
  size_t off = 0;
  auto alloc = [&](size_t bytes) -> char* {
    char* p = ws + off;
    off += (bytes + 255) & ~(size_t)255;
    return p;
  };
  bf16_t* xb    = (bf16_t*)alloc((size_t)Nrows * Dn * sizeof(bf16_t));
  bf16_t* Wqkvb = (bf16_t*)alloc((size_t)3 * Dn * Dn * sizeof(bf16_t));
  bf16_t* Wob   = (bf16_t*)alloc((size_t)Dn * Dn * sizeof(bf16_t));
  bf16_t* qb    = (bf16_t*)alloc((size_t)Bn * Hn * Tn * DHn * sizeof(bf16_t));
  bf16_t* kb    = (bf16_t*)alloc((size_t)Bn * Hn * Tn * DHn * sizeof(bf16_t));
  bf16_t* vtb   = (bf16_t*)alloc((size_t)Bn * Hn * DHn * Tn * sizeof(bf16_t));
  bf16_t* attnb = (bf16_t*)alloc((size_t)Nrows * Dn * sizeof(bf16_t));

  int nx = Nrows * Dn;
  f32_to_bf16_kernel<<<(nx + 255) / 256, 256, 0, stream>>>(x, xb, nx);
  int nw = 3 * Dn * Dn;
  f32_to_bf16_kernel<<<(nw + 255) / 256, 256, 0, stream>>>(Wqkv, Wqkvb, nw);
  int no = Dn * Dn;
  f32_to_bf16_kernel<<<(no + 255) / 256, 256, 0, stream>>>(Wo, Wob, no);

  dim3 g1(3 * Dn / 64, Nrows / 128);
  gemm_wmma<1><<<g1, 256, 0, stream>>>(xb, Wqkvb, nullptr, qb, kb, vtb,
                                       Nrows, 3 * Dn, Dn);

  int nr = Bn * Hn * Tn * 32;
  rope_kernel<<<nr / 256, 256, 0, stream>>>(qb, kb);

  flash_attn_wmma<<<(Bn * Hn * (Tn / 16)) / 8, 256, 0, stream>>>(qb, kb, vtb, attnb);

  dim3 g2(Dn / 64, Nrows / 128);
  gemm_wmma<0><<<g2, 256, 0, stream>>>(attnb, Wob, out, nullptr, nullptr, nullptr,
                                       Nrows, Dn, Dn);
}
